// Attention_38886633898597
// MI455X (gfx1250) — matrix operand order repaired and checked by reading, not compiled, not run on hardware
//
#include <hip/hip_runtime.h>

// MI455X / gfx1250, wave32. All heavy matmuls via v_wmma_f32_16x16x32_f16.
// Pipeline:
//   K0: one-time f32 -> f16 conversion of x, Wq, Wkv, Wo (hoists all cvt out
//       of GEMM inner loops; those become pure b128-load -> wmma)
//   K1: qkv = x @ [Wq;Wkv]^T            (WMMA, f16 in/out, 16x64 wave tile)
//   K2: RoPE(q,k) + rearrange to [B,H,N,DH] f16, V -> [B,H,DH,N] f16
//   K3: fused two-pass mixed-head attention (WMMA + LDS head exchange)
//   K4: out = att @ Wo^T + bo           (WMMA, f32 out, 16x64 wave tile)
// Workspace: ~66 MB.

typedef __attribute__((ext_vector_type(16))) _Float16 v16h;
typedef __attribute__((ext_vector_type(8)))  _Float16 v8h;
typedef __attribute__((ext_vector_type(8)))  float    v8f;

#define B_   4
#define N_   2048
#define DIM_ 512
#define H_   8
#define DH_  64

__device__ __forceinline__ v8f wmma_f16(v16h a, v16h b, v8f c) {
  return __builtin_amdgcn_wmma_f32_16x16x32_f16(
      /*neg_a=*/false, a, /*neg_b=*/false, b,
      /*c_mod=*/(short)0, c, /*reuse_a=*/false, /*reuse_b=*/false);
}

// ---- fragment loaders (layouts per CDNA5 ISA §7.12.2) ----

// A 16x32 f16: lanes 0-15: M=lane, K in {0..7}u{16..23}; lanes 16-31: M=lane-16,
// K in {8..15}u{24..31}. base already offset to (m0,k0) element.
__device__ __forceinline__ v16h load_a_frag_f16(const _Float16* base, int rs) {
  int lane = threadIdx.x & 31;
  int mrow = lane & 15;
  int k0 = (lane < 16) ? 0 : 8;
  const _Float16* p = base + (size_t)mrow * rs;
  v8h lo = *(const v8h*)(p + k0);
  v8h hi = *(const v8h*)(p + 16 + k0);
  v16h a;
#pragma unroll
  for (int e = 0; e < 8; ++e) { a[e] = lo[e]; a[8 + e] = hi[e]; }
  return a;
}

// B 32x16 f16 with B[k][n] = row-major src[n][k]: lane = N column (mod 16),
// lanes 0-15 hold K {0..7}u{16..23}, lanes 16-31 hold K {8..15}u{24..31}.
__device__ __forceinline__ v16h load_b_frag_f16(const _Float16* base, int rs) {
  int lane = threadIdx.x & 31;
  int n = lane & 15;
  int kb = (lane < 16) ? 0 : 8;
  const _Float16* p = base + (size_t)n * rs;
  v8h lo = *(const v8h*)(p + kb);
  v8h hi = *(const v8h*)(p + 16 + kb);
  v16h b;
#pragma unroll
  for (int e = 0; e < 8; ++e) { b[e] = lo[e]; b[8 + e] = hi[e]; }
  return b;
}

// ---- K0: bulk f32 -> f16 conversion (grid covers exactly n elements) --------
__global__ void cvt_f32_to_f16_kernel(const float* __restrict__ src,
                                      _Float16* __restrict__ dst) {
  int idx = blockIdx.x * blockDim.x + threadIdx.x;
  dst[idx] = (_Float16)src[idx];
}

// ---- K1: fused QKV projection ----------------------------------------------
// qkv[m, f] = sum_d x[m,d] * W[f,d], f in [0,1536): Wq rows then Wkv rows.
// One wave computes a 16x64 output tile; inner loop is pure b128 loads + WMMA.
__global__ void qkv_gemm_kernel(const _Float16* __restrict__ x16,
                                const _Float16* __restrict__ wq16,
                                const _Float16* __restrict__ wkv16,
                                _Float16* __restrict__ qkv) {
  const int FT = (3 * DIM_) / 64;                 // 24 feature tiles of 64
  int tile = blockIdx.x * 8 + (threadIdx.x >> 5); // one 16x64 tile per wave
  int m0 = (tile / FT) * 16;
  int f0 = (tile % FT) * 64;                      // 64 | 512, never straddles
  const _Float16* w; int wr;
  if (f0 < DIM_) { w = wq16; wr = f0; } else { w = wkv16; wr = f0 - DIM_; }
  v8f c[4] = {};
  for (int kk = 0; kk < DIM_; kk += 32) {
    v16h a = load_a_frag_f16(x16 + (size_t)m0 * DIM_ + kk, DIM_);
#pragma unroll
    for (int t = 0; t < 4; ++t) {
      v16h b = load_b_frag_f16(w + (size_t)(wr + t * 16) * DIM_ + kk, DIM_);
      c[t] = wmma_f16(a, b, c[t]);
    }
  }
  int lane = threadIdx.x & 31;
  int col = lane & 15;
  int rb = (lane < 16) ? 0 : 8;
#pragma unroll
  for (int t = 0; t < 4; ++t)
#pragma unroll
    for (int r = 0; r < 8; ++r)
      qkv[(size_t)(m0 + rb + r) * (3 * DIM_) + f0 + t * 16 + col] =
          (_Float16)c[t][r];
}

// ---- K2: RoPE + rearrange ---------------------------------------------------
// q'[j<32]  = q[j]*cos(n*invf[j])    - q[2j+1]   * sin(n*invf[j])
// q'[j>=32] = q[j]*cos(n*invf[j-32]) + q[2(j-32)]* sin(n*invf[j-32])
__global__ void rope_arrange_kernel(const _Float16* __restrict__ qkv,
                                    _Float16* __restrict__ q16,
                                    _Float16* __restrict__ k16,
                                    _Float16* __restrict__ vt16) {
  int idx = blockIdx.x * blockDim.x + threadIdx.x;  // 2^22 threads exactly
  int d = idx & 63;
  int h = (idx >> 6) & 7;
  int n = (idx >> 9) & (N_ - 1);
  int b = idx >> 20;
  size_t rowr = ((size_t)b * N_ + n) * (3 * DIM_);
  int cb = h * 64;
  int fi = (d < 32) ? d : (d - 32);
  // inv_freq = 10000^(-fi/32) ; ln(10000)/32 = 0.28782313662425572
  float ang = (float)n * __expf(-(float)fi * 0.28782313662425572f);
  float sn, cs;
  __sincosf(ang, &sn, &cs);
  int pd = (d < 32) ? (2 * d + 1) : (2 * (d - 32));
  float sgn = (d < 32) ? -1.0f : 1.0f;

  float qv = (float)qkv[rowr + cb + d];
  float qp = (float)qkv[rowr + cb + pd];
  float kv = (float)qkv[rowr + DIM_ + cb + d];
  float kp = (float)qkv[rowr + DIM_ + cb + pd];
  float vv = (float)qkv[rowr + 2 * DIM_ + cb + d];

  size_t bh = (size_t)b * H_ + h;
  q16[(bh * N_ + n) * DH_ + d] = (_Float16)(qv * cs + sgn * qp * sn);
  k16[(bh * N_ + n) * DH_ + d] = (_Float16)(kv * cs + sgn * kp * sn);
  vt16[(bh * DH_ + d) * N_ + n] = (_Float16)vv;   // transposed for P@V B-frags
}

// ---- K3: fused two-pass mixed-head attention --------------------------------
// Block = 8 waves; wave g owns head g for one (batch, 16-query) tile.
__global__ void attn_kernel(const _Float16* __restrict__ q16,
                            const _Float16* __restrict__ k16,
                            const _Float16* __restrict__ vt16,
                            const float* __restrict__ Wpre,
                            const float* __restrict__ Wpost,
                            _Float16* __restrict__ att16) {
  __shared__ float Sbuf[H_][16][33];               // 16.9 KB, padded stride

  int b = blockIdx.x >> 7;                         // N_/16 = 128 tiles per batch
  int i0 = (blockIdx.x & 127) * 16;
  int g = threadIdx.x >> 5;
  int lane = threadIdx.x & 31;
  size_t bh = (size_t)b * H_ + g;

  float wpre[H_], wpost[H_];
#pragma unroll
  for (int h = 0; h < H_; ++h) {
    wpre[h]  = Wpre[g * H_ + h];
    wpost[h] = Wpost[g * H_ + h];
  }

  const _Float16* qbase = q16 + (bh * N_ + i0) * DH_;
  v16h aq0 = load_a_frag_f16(qbase + 0,  DH_);     // d 0..31
  v16h aq1 = load_a_frag_f16(qbase + 32, DH_);     // d 32..63

  int rhalf = (lane < 16) ? 0 : 8;                 // C-layout row half
  int col = lane & 15;
  const float scale = 0.125f;                      // DH^-0.5

  float mrun[8], lrun[8];
#pragma unroll
  for (int r = 0; r < 8; ++r) { mrun[r] = -1e30f; lrun[r] = 0.0f; }

  // ---------------- pass 1: softmax statistics over mixed logits -----------
  for (int j0 = 0; j0 < N_; j0 += 32) {
    const _Float16* kb = k16 + (bh * N_ + j0) * DH_;
    if (j0 + 32 < N_)
      __builtin_prefetch(k16 + (bh * N_ + j0 + 32) * DH_, 0, 1);
    v8f s[2];
#pragma unroll
    for (int js = 0; js < 2; ++js) {
      v16h bk0 = load_b_frag_f16(kb + (size_t)js * 16 * DH_ + 0,  DH_);
      v16h bk1 = load_b_frag_f16(kb + (size_t)js * 16 * DH_ + 32, DH_);
      v8f a = {};
      a = wmma_f16(aq0, bk0, a);
      a = wmma_f16(aq1, bk1, a);
      s[js] = a;
    }
#pragma unroll
    for (int js = 0; js < 2; ++js)
#pragma unroll
      for (int r = 0; r < 8; ++r)
        Sbuf[g][rhalf + r][js * 16 + col] = s[js][r] * scale;
    __syncthreads();
#pragma unroll
    for (int r = 0; r < 8; ++r) {
      float v0 = 0.0f, v1 = 0.0f;
#pragma unroll
      for (int h = 0; h < H_; ++h) {
        v0 += wpre[h] * Sbuf[h][rhalf + r][col];
        v1 += wpre[h] * Sbuf[h][rhalf + r][16 + col];
      }
      float mx = fmaxf(v0, v1);
#pragma unroll
      for (int off = 8; off >= 1; off >>= 1)
        mx = fmaxf(mx, __shfl_xor(mx, off, 32));   // stays in 16-lane half
      float mnew = fmaxf(mrun[r], mx);
      float e = __expf(v0 - mnew) + __expf(v1 - mnew);
#pragma unroll
      for (int off = 8; off >= 1; off >>= 1)
        e += __shfl_xor(e, off, 32);
      lrun[r] = lrun[r] * __expf(mrun[r] - mnew) + e;
      mrun[r] = mnew;
    }
    __syncthreads();
  }

  float invl[8];
#pragma unroll
  for (int r = 0; r < 8; ++r) invl[r] = 1.0f / lrun[r];

  // ---------------- pass 2: normalized P, Wpost mix, P@V -------------------
  v8f acc[4] = {};
  int mrow = lane & 15;                            // A-layout row
  int ka0 = (lane < 16) ? 0 : 8;                   // A-layout K base

  for (int j0 = 0; j0 < N_; j0 += 32) {
    const _Float16* kb = k16 + (bh * N_ + j0) * DH_;
    v8f s[2];
#pragma unroll
    for (int js = 0; js < 2; ++js) {
      v16h bk0 = load_b_frag_f16(kb + (size_t)js * 16 * DH_ + 0,  DH_);
      v16h bk1 = load_b_frag_f16(kb + (size_t)js * 16 * DH_ + 32, DH_);
      v8f a = {};
      a = wmma_f16(aq0, bk0, a);
      a = wmma_f16(aq1, bk1, a);
      s[js] = a;
    }
#pragma unroll
    for (int js = 0; js < 2; ++js)
#pragma unroll
      for (int r = 0; r < 8; ++r)
        Sbuf[g][rhalf + r][js * 16 + col] = s[js][r] * scale;
    __syncthreads();

    float pv[2][8];
#pragma unroll
    for (int r = 0; r < 8; ++r) {
      float v0 = 0.0f, v1 = 0.0f;
#pragma unroll
      for (int h = 0; h < H_; ++h) {
        v0 += wpre[h] * Sbuf[h][rhalf + r][col];
        v1 += wpre[h] * Sbuf[h][rhalf + r][16 + col];
      }
      pv[0][r] = __expf(v0 - mrun[r]) * invl[r];
      pv[1][r] = __expf(v1 - mrun[r]) * invl[r];
    }
    __syncthreads();
#pragma unroll
    for (int js = 0; js < 2; ++js)
#pragma unroll
      for (int r = 0; r < 8; ++r)
        Sbuf[g][rhalf + r][js * 16 + col] = pv[js][r];
    __syncthreads();

    // build Ptilde[g] = sum_h Wpost[g,h] * P[h] directly in A-fragment layout
    v16h pa;
#pragma unroll
    for (int e = 0; e < 8; ++e) {
      float t0 = 0.0f, t1 = 0.0f;
#pragma unroll
      for (int h = 0; h < H_; ++h) {
        t0 += wpost[h] * Sbuf[h][mrow][ka0 + e];
        t1 += wpost[h] * Sbuf[h][mrow][16 + ka0 + e];
      }
      pa[e]     = (_Float16)t0;
      pa[8 + e] = (_Float16)t1;
    }

    const _Float16* vb = vt16 + bh * DH_ * N_ + j0;
#pragma unroll
    for (int ds = 0; ds < 4; ++ds) {
      v16h bv = load_b_frag_f16(vb + (size_t)ds * 16 * N_, N_);
      acc[ds] = wmma_f16(pa, bv, acc[ds]);
    }
    __syncthreads();
  }

  // store head-major [b, n, g*64+d] f16 for the output projection
#pragma unroll
  for (int ds = 0; ds < 4; ++ds)
#pragma unroll
    for (int r = 0; r < 8; ++r) {
      int i = i0 + rhalf + r;
      int c = g * 64 + ds * 16 + col;
      att16[((size_t)b * N_ + i) * DIM_ + c] = (_Float16)acc[ds][r];
    }
}

// ---- K4: output projection + bias (16x64 wave tile) -------------------------
__global__ void out_gemm_kernel(const _Float16* __restrict__ att16,
                                const _Float16* __restrict__ wo16,
                                const float* __restrict__ bo,
                                float* __restrict__ out) {
  const int FT = DIM_ / 64;                        // 8 feature tiles of 64
  int tile = blockIdx.x * 8 + (threadIdx.x >> 5);
  int m0 = (tile / FT) * 16;
  int f0 = (tile % FT) * 64;
  v8f c[4] = {};
  for (int kk = 0; kk < DIM_; kk += 32) {
    v16h a = load_a_frag_f16(att16 + (size_t)m0 * DIM_ + kk, DIM_);
#pragma unroll
    for (int t = 0; t < 4; ++t) {
      v16h b = load_b_frag_f16(wo16 + (size_t)(f0 + t * 16) * DIM_ + kk, DIM_);
      c[t] = wmma_f16(a, b, c[t]);
    }
  }
  int lane = threadIdx.x & 31;
  int col = lane & 15;
  int rb = (lane < 16) ? 0 : 8;
#pragma unroll
  for (int t = 0; t < 4; ++t) {
    float bias = bo[f0 + t * 16 + col];
#pragma unroll
    for (int r = 0; r < 8; ++r)
      out[(size_t)(m0 + rb + r) * DIM_ + f0 + t * 16 + col] = c[t][r] + bias;
  }
}

// ---- host glue --------------------------------------------------------------
extern "C" void kernel_launch(void* const* d_in, const int* in_sizes, int n_in,
                              void* d_out, int out_size, void* d_ws, size_t ws_size,
                              hipStream_t stream) {
  (void)in_sizes; (void)n_in; (void)out_size; (void)ws_size;
  const float* x     = (const float*)d_in[0];
  const float* Wq    = (const float*)d_in[1];
  const float* Wkv   = (const float*)d_in[2];
  const float* Wo    = (const float*)d_in[3];
  const float* bo    = (const float*)d_in[4];
  const float* Wpre  = (const float*)d_in[5];
  const float* Wpost = (const float*)d_in[6];
  float* out = (float*)d_out;

  // workspace carve-up (f16 buffers, ~66 MB total)
  char* ws = (char*)d_ws;
  _Float16* qkv  = (_Float16*)(ws);                // 8192*1536*2 = 24 MB
  _Float16* q16  = (_Float16*)(ws + 25165824);     //  8 MB
  _Float16* k16  = (_Float16*)(ws + 33554432);     //  8 MB
  _Float16* vt16 = (_Float16*)(ws + 41943040);     //  8 MB
  _Float16* att  = (_Float16*)(ws + 50331648);     //  8 MB
  _Float16* x16  = (_Float16*)(ws + 58720256);     //  8 MB (8192*512)
  _Float16* wq16 = (_Float16*)(ws + 67108864);     //  0.5 MB (512*512)
  _Float16* wkv16= (_Float16*)(ws + 67633152);     //  1 MB (1024*512)
  _Float16* wo16 = (_Float16*)(ws + 68681728);     //  0.5 MB (512*512)

  // K0: one-time f32 -> f16 conversions (exact-cover grids)
  cvt_f32_to_f16_kernel<<<16384, 256, 0, stream>>>(x, x16);     // 2^22 elems
  cvt_f32_to_f16_kernel<<<1024, 256, 0, stream>>>(Wq, wq16);    // 2^18 elems
  cvt_f32_to_f16_kernel<<<2048, 256, 0, stream>>>(Wkv, wkv16);  // 2^19 elems
  cvt_f32_to_f16_kernel<<<1024, 256, 0, stream>>>(Wo, wo16);    // 2^18 elems

  // K1: 512 M-tiles * 24 F64-tiles, 8 wave-tiles per block
  qkv_gemm_kernel<<<1536, 256, 0, stream>>>(x16, wq16, wkv16, qkv);
  // K2: B*N*DIM = 2^22 elementwise threads
  rope_arrange_kernel<<<16384, 256, 0, stream>>>(qkv, q16, k16, vt16);
  // K3: B * (N/16) query tiles, 8 waves (= 8 heads) per block
  attn_kernel<<<512, 256, 0, stream>>>(q16, k16, vt16, Wpre, Wpost, att);
  // K4: 512 M-tiles * 8 F64-tiles
  out_gemm_kernel<<<512, 256, 0, stream>>>(att, wo16, bo, out);
}
